// LinearGLUMoELayer_29600914604410
// MI455X (gfx1250) — matrix operand order repaired and checked by reading, not compiled, not run on hardware
//
#include <hip/hip_runtime.h>
#include <hip/hip_bf16.h>

#define D_MODEL 2048
#define H_E     1024
#define N_EXP   8
#define N_TOK   4096

typedef __bf16 bf16x16 __attribute__((ext_vector_type(16)));
typedef __bf16 bf16x8  __attribute__((ext_vector_type(8)));
typedef float  f32x8   __attribute__((ext_vector_type(8)));

__device__ __forceinline__ unsigned short f32_to_bf16_rne(float f) {
  unsigned int u = __float_as_uint(f);
  u += 0x7FFFu + ((u >> 16) & 1u);
  return (unsigned short)(u >> 16);
}

// A-matrix frag (16x32 bf16): per lane 16 contiguous K values (two 8-elem chunks at K+0 and K+8)
// (lanes>=16 shifted by +16 in K). p already includes row base + laneHiA.
__device__ __forceinline__ bf16x16 load_frag_a(const __bf16* p) {
  bf16x8 lo = *(const bf16x8*)(p);
  bf16x8 hi = *(const bf16x8*)(p + 8);
  return __builtin_shufflevector(lo, hi, 0,1,2,3,4,5,6,7,8,9,10,11,12,13,14,15);
}

// B-matrix frag (32x16 bf16): column in lane (lane&15), per lane 16
// contiguous K values (lanes<16: K 0..15, lanes>=16: K 16..31).
// p already includes column base + laneHiB; 32B aligned.
__device__ __forceinline__ bf16x16 load_frag_b(const __bf16* p) {
  return *(const bf16x16*)(p);
}

// ---------------- fp32 -> bf16 conversion (vector of 4) ----------------
__global__ __launch_bounds__(256)
void cvt_f32_to_bf16(const float* __restrict__ src, unsigned short* __restrict__ dst, int n4) {
  int i = blockIdx.x * blockDim.x + threadIdx.x;
  int stride = gridDim.x * blockDim.x;
  for (; i < n4; i += stride) {
    float4 v = ((const float4*)src)[i];
    unsigned int lo = (unsigned int)f32_to_bf16_rne(v.x) | ((unsigned int)f32_to_bf16_rne(v.y) << 16);
    unsigned int hi = (unsigned int)f32_to_bf16_rne(v.z) | ((unsigned int)f32_to_bf16_rne(v.w) << 16);
    ((uint2*)dst)[i] = make_uint2(lo, hi);
  }
}

// ---------------- gate: tanh(x@Wg1^T)@Wg2^T, top-2, softmax ----------------
__global__ __launch_bounds__(256)
void gate_kernel(const float* __restrict__ x, const float* __restrict__ Wg1,
                 const float* __restrict__ Wg2, int* __restrict__ cnt,
                 int* __restrict__ route_e, float* __restrict__ route_s) {
  int tok  = (blockIdx.x * blockDim.x + threadIdx.x) >> 5;
  int lane = threadIdx.x & 31;
  if (tok >= N_TOK) return;
  const float* xr = x + (size_t)tok * D_MODEL;

  float acc[N_EXP];
#pragma unroll
  for (int e = 0; e < N_EXP; ++e) acc[e] = 0.f;
  for (int d = lane * 4; d < D_MODEL; d += 32 * 4) {
    float4 xv = *(const float4*)(xr + d);
#pragma unroll
    for (int e = 0; e < N_EXP; ++e) {
      float4 wv = *(const float4*)(Wg1 + e * D_MODEL + d);
      acc[e] += xv.x * wv.x + xv.y * wv.y + xv.z * wv.z + xv.w * wv.w;
    }
  }
#pragma unroll
  for (int e = 0; e < N_EXP; ++e) {
#pragma unroll
    for (int off = 16; off > 0; off >>= 1)
      acc[e] += __shfl_xor(acc[e], off, 32);
  }
  if (lane == 0) {
    float a[N_EXP], l[N_EXP];
#pragma unroll
    for (int e = 0; e < N_EXP; ++e) a[e] = tanhf(acc[e]);
#pragma unroll
    for (int j = 0; j < N_EXP; ++j) {
      float s = 0.f;
#pragma unroll
      for (int i = 0; i < N_EXP; ++i) s += Wg2[j * N_EXP + i] * a[i];
      l[j] = s;
    }
    int i0 = 0;
#pragma unroll
    for (int j = 1; j < N_EXP; ++j) if (l[j] > l[i0]) i0 = j;
    int i1 = (i0 == 0) ? 1 : 0;
#pragma unroll
    for (int j = 0; j < N_EXP; ++j) if (j != i0 && l[j] > l[i1]) i1 = j;
    float s0 = 1.f / (1.f + __expf(l[i1] - l[i0]));  // softmax over {l[i0], l[i1]}
    float s1 = 1.f - s0;
    route_e[2 * tok] = i0;  route_e[2 * tok + 1] = i1;
    route_s[2 * tok] = s0;  route_s[2 * tok + 1] = s1;
    atomicAdd(&cnt[i0], 1);
    atomicAdd(&cnt[i1], 1);
  }
}

__global__ void offsets_kernel(const int* __restrict__ cnt, int* __restrict__ offsets) {
  if (blockIdx.x == 0 && threadIdx.x == 0) {
    int o = 0;
    for (int e = 0; e < N_EXP; ++e) { offsets[e] = o; o += cnt[e]; }
    offsets[N_EXP] = o;
  }
}

__global__ __launch_bounds__(256)
void assign_kernel(const int* __restrict__ route_e, const int* __restrict__ offsets,
                   int* __restrict__ fill, int* __restrict__ assign_tok,
                   int* __restrict__ route_pos) {
  int t = blockIdx.x * blockDim.x + threadIdx.x;
  if (t >= N_TOK) return;
#pragma unroll
  for (int k = 0; k < 2; ++k) {
    int e = route_e[2 * t + k];
    int pos = offsets[e] + atomicAdd(&fill[e], 1);
    assign_tok[pos] = t;
    route_pos[2 * t + k] = pos;
  }
}

// ---------------- up GEMM: h = silu(Xg @ Wup[e]^T + bup[e]) ----------------
// block = 128 thr (4 waves); block tile M128 x N64; wave w: rows [32w,32w+32) x 64 cols
// per K=32 step: 2 A frags + 4 B frags -> 8 WMMAs
__global__ __launch_bounds__(128)
void up_gemm(const unsigned short* __restrict__ xbf,
             const unsigned short* __restrict__ wup,
             const float* __restrict__ bup,
             const int* __restrict__ assign_tok,
             const int* __restrict__ offsets,
             unsigned short* __restrict__ hbuf) {
  const int e    = blockIdx.z;
  const int aoff = offsets[e];
  const int Ne   = offsets[e + 1] - aoff;
  const int m_base = blockIdx.y * 128;
  if (m_base >= Ne) return;
  const int n_base  = blockIdx.x * 64;
  const int lane    = threadIdx.x & 31;
  const int wv      = threadIdx.x >> 5;
  const int laneHiA = (lane >> 4) << 4;   // +16 bf16 in K for upper half-wave
  const int laneHiB = (lane >> 4) << 4;   // +16 bf16 in K for upper half-wave
  const int lrow    = lane & 15;
  const int wbaseM  = m_base + wv * 32;

  int r0 = wbaseM + lrow;
  int r1 = wbaseM + 16 + lrow;
  int r0c = r0 < Ne ? r0 : (Ne - 1);
  int r1c = r1 < Ne ? r1 : (Ne - 1);
  const __bf16* aptr0 = (const __bf16*)xbf + (size_t)assign_tok[aoff + r0c] * D_MODEL + laneHiA;
  const __bf16* aptr1 = (const __bf16*)xbf + (size_t)assign_tok[aoff + r1c] * D_MODEL + laneHiA;
  const __bf16* wbase = (const __bf16*)wup + (size_t)e * H_E * D_MODEL;
  const __bf16* bptr[4];
#pragma unroll
  for (int f = 0; f < 4; ++f)
    bptr[f] = wbase + (size_t)(n_base + f * 16 + lrow) * D_MODEL + laneHiB;

  f32x8 acc0[4] = {};
  f32x8 acc1[4] = {};
  for (int kk = 0; kk < D_MODEL; kk += 32) {
    __builtin_prefetch(aptr0 + kk + 256, 0, 1);
    __builtin_prefetch(aptr1 + kk + 256, 0, 1);
    bf16x16 a0 = load_frag_a(aptr0 + kk);
    bf16x16 a1 = load_frag_a(aptr1 + kk);
#pragma unroll
    for (int f = 0; f < 4; ++f) {
      bf16x16 b = load_frag_b(bptr[f] + kk);
      acc0[f] = __builtin_amdgcn_wmma_f32_16x16x32_bf16(false, a0, false, b,
                                                        (short)0, acc0[f], false, false);
      acc1[f] = __builtin_amdgcn_wmma_f32_16x16x32_bf16(false, a1, false, b,
                                                        (short)0, acc1[f], false, false);
    }
  }

#pragma unroll
  for (int f = 0; f < 4; ++f) {
    int n = n_base + f * 16 + lrow;
    float bias = bup[e * H_E + n];
#pragma unroll
    for (int g = 0; g < 2; ++g) {
#pragma unroll
      for (int r = 0; r < 8; ++r) {
        int m = wbaseM + g * 16 + r + ((lane >> 4) << 3);
        if (m < Ne) {
          float u = (g == 0 ? acc0[f][r] : acc1[f][r]) + bias;
          float h = u * (1.0f / (1.0f + __expf(-u)));   // silu
          hbuf[(size_t)(aoff + m) * H_E + n] = f32_to_bf16_rne(h);
        }
      }
    }
  }
}

// ---------------- down GEMM: dbuf = h @ Wdown[e]^T + bdown[e] ----------------
__global__ __launch_bounds__(128)
void down_gemm(const unsigned short* __restrict__ hbuf,
               const unsigned short* __restrict__ wdn,
               const float* __restrict__ bdown,
               const int* __restrict__ offsets,
               float* __restrict__ dbuf) {
  const int e    = blockIdx.z;
  const int aoff = offsets[e];
  const int Ne   = offsets[e + 1] - aoff;
  const int m_base = blockIdx.y * 128;
  if (m_base >= Ne) return;
  const int n_base  = blockIdx.x * 64;
  const int lane    = threadIdx.x & 31;
  const int wv      = threadIdx.x >> 5;
  const int laneHiA = (lane >> 4) << 4;
  const int laneHiB = (lane >> 4) << 4;
  const int lrow    = lane & 15;
  const int wbaseM  = m_base + wv * 32;

  int r0 = wbaseM + lrow;
  int r1 = wbaseM + 16 + lrow;
  int r0c = r0 < Ne ? r0 : (Ne - 1);
  int r1c = r1 < Ne ? r1 : (Ne - 1);
  const __bf16* aptr0 = (const __bf16*)hbuf + (size_t)(aoff + r0c) * H_E + laneHiA;
  const __bf16* aptr1 = (const __bf16*)hbuf + (size_t)(aoff + r1c) * H_E + laneHiA;
  const __bf16* wbase = (const __bf16*)wdn + (size_t)e * D_MODEL * H_E;
  const __bf16* bptr[4];
#pragma unroll
  for (int f = 0; f < 4; ++f)
    bptr[f] = wbase + (size_t)(n_base + f * 16 + lrow) * H_E + laneHiB;

  f32x8 acc0[4] = {};
  f32x8 acc1[4] = {};
  for (int kk = 0; kk < H_E; kk += 32) {
    __builtin_prefetch(aptr0 + kk + 256, 0, 1);
    __builtin_prefetch(aptr1 + kk + 256, 0, 1);
    bf16x16 a0 = load_frag_a(aptr0 + kk);
    bf16x16 a1 = load_frag_a(aptr1 + kk);
#pragma unroll
    for (int f = 0; f < 4; ++f) {
      bf16x16 b = load_frag_b(bptr[f] + kk);
      acc0[f] = __builtin_amdgcn_wmma_f32_16x16x32_bf16(false, a0, false, b,
                                                        (short)0, acc0[f], false, false);
      acc1[f] = __builtin_amdgcn_wmma_f32_16x16x32_bf16(false, a1, false, b,
                                                        (short)0, acc1[f], false, false);
    }
  }

#pragma unroll
  for (int f = 0; f < 4; ++f) {
    int n = n_base + f * 16 + lrow;
    float bias = bdown[e * D_MODEL + n];
#pragma unroll
    for (int g = 0; g < 2; ++g) {
#pragma unroll
      for (int r = 0; r < 8; ++r) {
        int m = wbaseM + g * 16 + r + ((lane >> 4) << 3);
        if (m < Ne)
          dbuf[(size_t)(aoff + m) * D_MODEL + n] = (g == 0 ? acc0[f][r] : acc1[f][r]) + bias;
      }
    }
  }
}

// ---------------- combine: out[t] = s0*down[pos0] + s1*down[pos1] ----------------
__global__ __launch_bounds__(256)
void combine_kernel(const float* __restrict__ dbuf, const float* __restrict__ route_s,
                    const int* __restrict__ route_pos, float* __restrict__ out) {
  int t = blockIdx.x;
  float s0 = route_s[2 * t], s1 = route_s[2 * t + 1];
  size_t p0 = (size_t)route_pos[2 * t] * D_MODEL;
  size_t p1 = (size_t)route_pos[2 * t + 1] * D_MODEL;
  const float4* a = (const float4*)(dbuf + p0);
  const float4* b = (const float4*)(dbuf + p1);
  float4* o = (float4*)(out + (size_t)t * D_MODEL);
  for (int i = threadIdx.x; i < D_MODEL / 4; i += blockDim.x) {
    float4 va = a[i], vb = b[i];
    float4 r;
    r.x = s0 * va.x + s1 * vb.x;
    r.y = s0 * va.y + s1 * vb.y;
    r.z = s0 * va.z + s1 * vb.z;
    r.w = s0 * va.w + s1 * vb.w;
    o[i] = r;
  }
}

extern "C" void kernel_launch(void* const* d_in, const int* in_sizes, int n_in,
                              void* d_out, int out_size, void* d_ws, size_t ws_size,
                              hipStream_t stream) {
  const float* x     = (const float*)d_in[0];
  const float* Wg1   = (const float*)d_in[1];
  const float* Wg2   = (const float*)d_in[2];
  const float* Wup   = (const float*)d_in[3];
  const float* bup   = (const float*)d_in[4];
  const float* Wdown = (const float*)d_in[5];
  const float* bdown = (const float*)d_in[6];
  float* out = (float*)d_out;

  char* ws = (char*)d_ws;
  size_t off = 0;
  auto alloc = [&](size_t bytes) -> char* {
    char* p = ws + off;
    off += (bytes + 255) & ~(size_t)255;
    return p;
  };

  int*   cnt        = (int*)alloc(16 * sizeof(int));       // cnt[8] + fill[8]
  int*   fill       = cnt + 8;
  int*   offsets    = (int*)alloc(16 * sizeof(int));
  int*   route_e    = (int*)alloc((size_t)2 * N_TOK * sizeof(int));
  float* route_s    = (float*)alloc((size_t)2 * N_TOK * sizeof(float));
  int*   route_pos  = (int*)alloc((size_t)2 * N_TOK * sizeof(int));
  int*   assign_tok = (int*)alloc((size_t)2 * N_TOK * sizeof(int));
  unsigned short* xbf   = (unsigned short*)alloc((size_t)N_TOK * D_MODEL * 2);
  unsigned short* wup16 = (unsigned short*)alloc((size_t)N_EXP * H_E * D_MODEL * 2);
  unsigned short* wdn16 = (unsigned short*)alloc((size_t)N_EXP * D_MODEL * H_E * 2);
  unsigned short* hbuf  = (unsigned short*)alloc((size_t)2 * N_TOK * H_E * 2);
  float* dbuf           = (float*)alloc((size_t)2 * N_TOK * D_MODEL * sizeof(float));
  (void)ws_size; (void)in_sizes; (void)n_in; (void)out_size;

  hipMemsetAsync(cnt, 0, 16 * sizeof(int), stream);

  // fp32 -> bf16 conversions
  {
    int n4x = N_TOK * D_MODEL / 4;
    int n4w = N_EXP * H_E * D_MODEL / 4;
    cvt_f32_to_bf16<<<2048, 256, 0, stream>>>(x, xbf, n4x);
    cvt_f32_to_bf16<<<4096, 256, 0, stream>>>(Wup, wup16, n4w);
    cvt_f32_to_bf16<<<4096, 256, 0, stream>>>(Wdown, wdn16, n4w);
  }

  gate_kernel<<<N_TOK / 8, 256, 0, stream>>>(x, Wg1, Wg2, cnt, route_e, route_s);
  offsets_kernel<<<1, 32, 0, stream>>>(cnt, offsets);
  assign_kernel<<<N_TOK / 256, 256, 0, stream>>>(route_e, offsets, fill, assign_tok, route_pos);

  up_gemm<<<dim3(H_E / 64, N_TOK / 128, N_EXP), 128, 0, stream>>>(
      xbf, wup16, bup, assign_tok, offsets, hbuf);
  down_gemm<<<dim3(D_MODEL / 64, N_TOK / 128, N_EXP), 128, 0, stream>>>(
      hbuf, wdn16, bdown, offsets, dbuf);
  combine_kernel<<<N_TOK, 256, 0, stream>>>(dbuf, route_s, route_pos, out);
}
